// GlobalPoint_6554120094164
// MI455X (gfx1250) — matrix operand order repaired and checked by reading, not compiled, not run on hardware
//
#include <hip/hip_runtime.h>
#include <math.h>

typedef __attribute__((ext_vector_type(16))) __bf16         v16bf;
typedef __attribute__((ext_vector_type(8)))  float          v8f;
typedef __attribute__((ext_vector_type(16))) unsigned short v16u;
typedef __attribute__((ext_vector_type(8)))  unsigned short v8u;

#define QK_ELEMS   (16 * 9 * 512 * 64)   // 4,718,592 per q/k plane
#define HID_ELEMS  (8192 * 768)          // 6,291,456
#define W_ELEMS    (768 * 1152)          // 884,736
#define TAB_ELEMS  (512 * 32)            // 16,384 per cos/sin table
#define NEGBIG     1.0e12f

__device__ __forceinline__ unsigned short f2bf(float f) {
    unsigned u = __float_as_uint(f);
    return (unsigned short)((u + 0x7FFFu + ((u >> 16) & 1u)) >> 16);  // RNE
}
__device__ __forceinline__ float bf2f(unsigned short h) {
    return __uint_as_float(((unsigned)h) << 16);
}
__device__ __forceinline__ v16u cat8(v8u lo, v8u hi) {
    return __builtin_shufflevector(lo, hi, 0,1,2,3,4,5,6,7,8,9,10,11,12,13,14,15);
}
__device__ __forceinline__ v8f wmma3(v16u ah, v16u al, v16u bh, v16u bl, v8f c) {
    v16bf Ah = __builtin_bit_cast(v16bf, ah);
    v16bf Al = __builtin_bit_cast(v16bf, al);
    v16bf Bh = __builtin_bit_cast(v16bf, bh);
    v16bf Bl = __builtin_bit_cast(v16bf, bl);
    c = __builtin_amdgcn_wmma_f32_16x16x32_bf16(false, Ah, false, Bh, (short)0, c, false, false);
    c = __builtin_amdgcn_wmma_f32_16x16x32_bf16(false, Ah, false, Bl, (short)0, c, false, false);
    c = __builtin_amdgcn_wmma_f32_16x16x32_bf16(false, Al, false, Bh, (short)0, c, false, false);
    return c;
}

// ---------------------------------------------------------------------------
// Prep 1: hidden (f32 [8192,768]) -> bf16 hi/lo planes, same layout
// ---------------------------------------------------------------------------
__global__ __launch_bounds__(256) void conv_hidden_kernel(
    const float* __restrict__ hidden,
    unsigned short* __restrict__ hh, unsigned short* __restrict__ hl)
{
    int idx = blockIdx.x * 256 + threadIdx.x;
    if (idx >= HID_ELEMS) return;
    float f = hidden[idx];
    unsigned short h = f2bf(f);
    hh[idx] = h;
    hl[idx] = f2bf(f - bf2f(h));
}

// ---------------------------------------------------------------------------
// Prep 2: W (f32 [768,1152]) -> transposed bf16 hi/lo planes wt[n*768+k]
// ---------------------------------------------------------------------------
__global__ __launch_bounds__(256) void conv_w_kernel(
    const float* __restrict__ W,
    unsigned short* __restrict__ wh, unsigned short* __restrict__ wl)
{
    int idx = blockIdx.x * 256 + threadIdx.x;
    if (idx >= W_ELEMS) return;
    int n = idx / 768, k = idx - n * 768;
    float f = W[(size_t)k * 1152 + n];
    unsigned short h = f2bf(f);
    wh[idx] = h;
    wl[idx] = f2bf(f - bf2f(h));
}

// ---------------------------------------------------------------------------
// Prep 3: RoPE tables  ct/st[s*32+i] = cos/sin(s * 10000^(-2i/64))
// ---------------------------------------------------------------------------
__global__ __launch_bounds__(256) void rope_table_kernel(
    float* __restrict__ ct, float* __restrict__ st)
{
    int idx = blockIdx.x * 256 + threadIdx.x;
    if (idx >= TAB_ELEMS) return;
    int s = idx >> 5, i = idx & 31;
    float inv = __powf(10000.0f, -2.0f * (float)i / 64.0f);
    float ang = (float)s * inv;
    ct[idx] = cosf(ang);
    st[idx] = sinf(ang);
}

// ---------------------------------------------------------------------------
// Kernel 1: proj = hidden @ W + b, RoPE, emit q/k bf16 hi/lo planes (B,T,S,D)
// One wave computes a 32x64 tile (2 M x 4 N accumulators); bf16x3 split.
// ---------------------------------------------------------------------------
__global__ __launch_bounds__(256) void proj_rope_kernel(
    const unsigned short* __restrict__ hh, const unsigned short* __restrict__ hl,
    const unsigned short* __restrict__ wh, const unsigned short* __restrict__ wl,
    const float* __restrict__ bias,
    const float* __restrict__ ct, const float* __restrict__ st,
    unsigned short* __restrict__ qh, unsigned short* __restrict__ ql,
    unsigned short* __restrict__ kh, unsigned short* __restrict__ kl)
{
    const int lane = threadIdx.x & 31;
    const int warp = threadIdx.x >> 5;
    const int tile = blockIdx.x * 8 + warp;     // 256 * 18 = 4608 wave-tiles
    const int tm = tile & 255;                  // 256 row tiles of 32 rows
    const int cg = tile >> 8;                   // 18 column groups of 64
    const int row0 = tm * 32, col0 = cg * 64;
    const int m0    = row0 + (lane & 15);
    const int m1    = m0 + 16;
    const int abase = (lane < 16) ? 0 : 8;
    const int bbase = (lane < 16) ? 0 : 8;

    const unsigned short* a0h = hh + (size_t)m0 * 768;
    const unsigned short* a0l = hl + (size_t)m0 * 768;
    const unsigned short* a1h = hh + (size_t)m1 * 768;
    const unsigned short* a1l = hl + (size_t)m1 * 768;

    v8f acc[2][4] = {};
    for (int kc = 0; kc < 768; kc += 32) {
        const int ao = kc + abase;
        v16u ah0 = cat8(*(const v8u*)(a0h + ao), *(const v8u*)(a0h + ao + 16));
        v16u al0 = cat8(*(const v8u*)(a0l + ao), *(const v8u*)(a0l + ao + 16));
        v16u ah1 = cat8(*(const v8u*)(a1h + ao), *(const v8u*)(a1h + ao + 16));
        v16u al1 = cat8(*(const v8u*)(a1l + ao), *(const v8u*)(a1l + ao + 16));
        #pragma unroll
        for (int u = 0; u < 4; ++u) {
            const size_t bo = (size_t)(col0 + u * 16 + (lane & 15)) * 768 + kc + bbase;
            v16u bh = cat8(*(const v8u*)(wh + bo), *(const v8u*)(wh + bo + 16));
            v16u bl = cat8(*(const v8u*)(wl + bo), *(const v8u*)(wl + bo + 16));
            acc[0][u] = wmma3(ah0, al0, bh, bl, acc[0][u]);
            acc[1][u] = wmma3(ah1, al1, bh, bl, acc[1][u]);
        }
    }

    // Epilogue: bias + RoPE via tables (pair partner = adjacent lane)
    #pragma unroll
    for (int r = 0; r < 2; ++r) {
        #pragma unroll
        for (int u = 0; u < 4; ++u) {
            const int  nb  = col0 + u * 16 + (lane & 15);
            const int  t   = nb >> 7;
            const int  j   = nb & 127;
            const int  jj  = j & 63;
            const bool isq = (j < 64);
            const int  ti  = jj >> 1;                 // table column
            const float sgn = (jj & 1) ? 1.0f : -1.0f;
            const float bval = bias[nb];
            unsigned short* __restrict__ dh = isq ? qh : kh;
            unsigned short* __restrict__ dl = isq ? ql : kl;
            #pragma unroll
            for (int v = 0; v < 8; ++v) {
                const int mm = row0 + r * 16 + v + ((lane < 16) ? 0 : 8);
                const int s  = mm & 511;
                const int bi = mm >> 9;
                float val = acc[r][u][v] + bval;
                float prt = __shfl_xor(val, 1, 32);
                float cs = ct[(s << 5) + ti];
                float sn = st[(s << 5) + ti];
                float rr = val * cs + sgn * prt * sn;
                size_t off = ((((size_t)bi * 9 + t) * 512 + s) << 6) + jj;
                unsigned short h = f2bf(rr);
                dh[off] = h;
                dl[off] = f2bf(rr - bf2f(h));
            }
        }
    }
}

// ---------------------------------------------------------------------------
// Kernel 2: logits[bt,m,n] = q[bt,m,:] . k[bt,n,:]  (bf16x3), mask + scale
// One wave computes a 32x64 logits tile (2 M x 4 N accumulators).
// ---------------------------------------------------------------------------
__global__ __launch_bounds__(256) void attn_logits_kernel(
    const unsigned short* __restrict__ qh, const unsigned short* __restrict__ ql,
    const unsigned short* __restrict__ kh, const unsigned short* __restrict__ kl,
    const int* __restrict__ mask, float* __restrict__ out)
{
    const int lane = threadIdx.x & 31;
    const int warp = threadIdx.x >> 5;
    const int tile = blockIdx.x * 8 + warp;     // 144 * 128 = 18432 wave-tiles
    const int bt = tile >> 7;                   // b*9 + t
    const int w  = tile & 127;
    const int m0 = (w >> 3) * 32;               // 16 row tiles of 32
    const int n0 = (w & 7) * 64;                // 8 column groups
    const size_t base = (size_t)bt * (512 * 64);
    const int am0   = m0 + (lane & 15);
    const int am1   = am0 + 16;
    const int abase = (lane < 16) ? 0 : 8;
    const int bbase = (lane < 16) ? 0 : 8;

    v8f acc[2][4] = {};
    #pragma unroll
    for (int d0 = 0; d0 < 64; d0 += 32) {
        const size_t ao0 = base + (size_t)am0 * 64 + d0 + abase;
        const size_t ao1 = base + (size_t)am1 * 64 + d0 + abase;
        v16u ah0 = cat8(*(const v8u*)(qh + ao0), *(const v8u*)(qh + ao0 + 16));
        v16u al0 = cat8(*(const v8u*)(ql + ao0), *(const v8u*)(ql + ao0 + 16));
        v16u ah1 = cat8(*(const v8u*)(qh + ao1), *(const v8u*)(qh + ao1 + 16));
        v16u al1 = cat8(*(const v8u*)(ql + ao1), *(const v8u*)(ql + ao1 + 16));
        #pragma unroll
        for (int u = 0; u < 4; ++u) {
            const size_t bo = base + (size_t)(n0 + u * 16 + (lane & 15)) * 64 + d0 + bbase;
            v16u bh = cat8(*(const v8u*)(kh + bo), *(const v8u*)(kh + bo + 16));
            v16u bl = cat8(*(const v8u*)(kl + bo), *(const v8u*)(kl + bo + 16));
            acc[0][u] = wmma3(ah0, al0, bh, bl, acc[0][u]);
            acc[1][u] = wmma3(ah1, al1, bh, bl, acc[1][u]);
        }
    }

    const int bidx = bt / 9;
    #pragma unroll
    for (int u = 0; u < 4; ++u) {
        const int bn = n0 + u * 16 + (lane & 15);
        const float pm = (float)mask[bidx * 512 + bn];
        const float negadd = (1.0f - pm) * NEGBIG;
        #pragma unroll
        for (int r = 0; r < 2; ++r) {
            #pragma unroll
            for (int v = 0; v < 8; ++v) {
                const int mr = m0 + r * 16 + v + ((lane < 16) ? 0 : 8);
                float val = acc[r][u][v] * pm - negadd;
                if (bn < mr) val -= NEGBIG;            // tril(.., -1): mask n < m
                out[((size_t)bt * 512 + mr) * 512 + bn] = val * 0.125f;
            }
        }
    }
}

extern "C" void kernel_launch(void* const* d_in, const int* in_sizes, int n_in,
                              void* d_out, int out_size, void* d_ws, size_t ws_size,
                              hipStream_t stream) {
    const float* hidden = (const float*)d_in[0];   // (16,512,768)
    const float* W      = (const float*)d_in[1];   // (768,1152)
    const float* bias   = (const float*)d_in[2];   // (1152,)
    const int*   mask   = (const int*)d_in[3];     // (16,512)
    float* out = (float*)d_out;                    // (16,9,512,512)

    unsigned short* qh = (unsigned short*)d_ws;
    unsigned short* ql = qh + QK_ELEMS;
    unsigned short* kh = ql + QK_ELEMS;
    unsigned short* kl = kh + QK_ELEMS;
    unsigned short* hh = kl + QK_ELEMS;
    unsigned short* hl = hh + HID_ELEMS;
    unsigned short* wh = hl + HID_ELEMS;
    unsigned short* wl = wh + W_ELEMS;
    float* ct = (float*)(wl + W_ELEMS);
    float* st = ct + TAB_ELEMS;

    conv_hidden_kernel<<<(HID_ELEMS + 255) / 256, 256, 0, stream>>>(hidden, hh, hl);
    conv_w_kernel<<<(W_ELEMS + 255) / 256, 256, 0, stream>>>(W, wh, wl);
    rope_table_kernel<<<(TAB_ELEMS + 255) / 256, 256, 0, stream>>>(ct, st);
    proj_rope_kernel<<<576, 256, 0, stream>>>(hh, hl, wh, wl, bias, ct, st, qh, ql, kh, kl);
    attn_logits_kernel<<<2304, 256, 0, stream>>>(qh, ql, kh, kl, mask, out);
}
